// DeformableConv2dSimple_43602507989499
// MI455X (gfx1250) — matrix operand order repaired and checked by reading, not compiled, not run on hardware
//
#include <hip/hip_runtime.h>
#include <hip/hip_bf16.h>
#include <math.h>

// ---------------- CDNA5 WMMA types ----------------
typedef __attribute__((ext_vector_type(16))) __bf16 v16bf;
typedef __attribute__((ext_vector_type(8)))  float  v8f;

// Problem constants (from reference setup_inputs)
#define BB   4
#define CC   128
#define HH   64
#define WW   64
#define OO   128
#define KT   9          // K*K taps
#define RR   (CC*KT)    // 1152 reduction length, r = k_tap*128 + c
#define NCH  36         // RR/32 K-chunks (chunk = k_tap*4 + c_blk)
#define OTIL 8          // OO/16 output-channel tiles
#define NT   2          // spatial N-tiles per block (32 columns)

__device__ __forceinline__ unsigned short f2bf(float f) {
    unsigned u = __float_as_uint(f);
    unsigned r = u + 0x7FFFu + ((u >> 16) & 1u);   // round-to-nearest-even
    return (unsigned short)(r >> 16);
}

// ------------------------------------------------------------------
// Kernel 1: offset(18ch) + mask(9ch, sigmoid) direct conv, fp32.
// offm layout: [b][27][64][64]
// ------------------------------------------------------------------
__global__ void offmask_kernel(const float* __restrict__ x,
                               const float* __restrict__ w_off,
                               const float* __restrict__ b_off,
                               const float* __restrict__ w_mask,
                               const float* __restrict__ b_mask,
                               float* __restrict__ offm) {
    int idx = blockIdx.x * 256 + threadIdx.x;
    const int total = BB * 27 * HH * WW;
    if (idx >= total) return;
    int w  = idx & 63;
    int h  = (idx >> 6) & 63;
    int ch = (idx >> 12) % 27;
    int b  = idx / (27 * HH * WW);

    const float* wt;
    float acc;
    if (ch < 18) { wt = w_off  + ch * CC * KT;        acc = b_off[ch]; }
    else         { wt = w_mask + (ch - 18) * CC * KT; acc = b_mask[ch - 18]; }

    for (int c = 0; c < CC; ++c) {
        const float* xc = x + (((size_t)b * CC + c) << 12);
        const float* wc = wt + c * KT;
        #pragma unroll
        for (int ki = 0; ki < 3; ++ki) {
            int yy = h - 1 + ki;
            if (yy < 0 || yy >= HH) continue;
            #pragma unroll
            for (int kj = 0; kj < 3; ++kj) {
                int xx = w - 1 + kj;
                if (xx < 0 || xx >= WW) continue;
                acc = fmaf(xc[(yy << 6) + xx], wc[ki * 3 + kj], acc);
            }
        }
    }
    if (ch >= 18) acc = 1.0f / (1.0f + __expf(-acc));
    offm[idx] = acc;
}

// ------------------------------------------------------------------
// Kernel 2: pack weight (O,C,3,3) fp32 -> bf16 A-fragments.
// Reduction index r = k_tap*128 + c ; chunk = r/32 = k_tap*4 + c_blk.
// A-matrix 16x32 (MxK) 16-bit layout (ISA 7.12.2):
//   lane g=lane/16, m=lane%16 ; VGPR v, half h:
//   v<4 : K = 2v + h + 8g     ; v>=4 : K = 2(v-4) + h + 16 + 8g
// wA dword index = ((chunk*8 + o_tile)*32 + lane)*8 + v
// ------------------------------------------------------------------
__global__ void packw_kernel(const float* __restrict__ weight,
                             unsigned int* __restrict__ wA) {
    int idx = blockIdx.x * 256 + threadIdx.x;          // one dword each
    const int total = NCH * OTIL * 32 * 8;             // 73728
    if (idx >= total) return;
    int v     = idx & 7;
    int lane  = (idx >> 3) & 31;
    int ot    = (idx >> 8) & 7;
    int chunk = idx >> 11;
    int k_tap = chunk >> 2;
    int c_blk = chunk & 3;
    int g     = lane >> 4;
    int o     = ot * 16 + (lane & 15);

    unsigned int pack = 0;
    #pragma unroll
    for (int hh = 0; hh < 2; ++hh) {
        int K = (v < 4) ? (2 * v + hh + 8 * g) : (2 * (v - 4) + hh + 16 + 8 * g);
        int c = c_blk * 32 + K;
        float wv = weight[((size_t)o * CC + c) * KT + k_tap];
        pack |= (unsigned int)f2bf(wv) << (16 * hh);
    }
    wA[idx] = pack;
}

// ------------------------------------------------------------------
// Kernel 3: deformable conv main. grid = (Wo/32, Ho, B), block = 256.
// Each wave owns one 16-o-channel tile and TWO spatial N-tiles
// (2 independent accumulators -> 2-way WMMA ILP, A-fragment reuse x2).
// B-matrix 32x16 (KxN) 16-bit layout: lane = N + 16*((K>>3)&1);
//   k'=K%8 + 8*(K>=16) -> VGPR v=k'/2, half=k'%2.  Stored pre-swizzled in LDS so
//   each lane reads one contiguous 32B run per (chunk, ntile).
// ------------------------------------------------------------------
__global__ void __launch_bounds__(256)
deform_wmma_kernel(const float* __restrict__ x,
                   const float* __restrict__ offm,
                   const unsigned short* __restrict__ wA,
                   const float* __restrict__ bias,
                   float* __restrict__ out) {
    __shared__ unsigned short sB[NCH * NT * 512];   // 72 KB of B fragments
    __shared__ float sPy[KT][32];
    __shared__ float sPx[KT][32];
    __shared__ float sM [KT][32];

    const int t  = threadIdx.x;
    const int w0 = blockIdx.x * 32;
    const int h  = blockIdx.y;
    const int b  = blockIdx.z;

    // ---- Phase 0: per-(tap, n) sample coordinates + mask (288 items) ----
    for (int i = t; i < KT * 32; i += 256) {
        int k = i >> 5, n = i & 31;
        int ki = k / 3, kj = k % 3;
        int w = w0 + n;
        size_t base = ((size_t)b * 27 * HH + h) * WW + w;
        float dy = offm[base + (size_t)(2 * k)     * HH * WW];
        float dx = offm[base + (size_t)(2 * k + 1) * HH * WW];
        float m  = offm[base + (size_t)(18 + k)    * HH * WW];
        sPy[k][n] = (float)(h - 1 + ki) + dy;
        sPx[k][n] = (float)(w - 1 + kj) + dx;
        sM [k][n] = m;
    }
    __syncthreads();

    // ---- Phase 1: bilinear sample -> bf16 -> B-fragment LDS ----
    for (int k = 0; k < KT; ++k) {
        #pragma unroll 2
        for (int it = 0; it < 16; ++it) {
            int idx = it * 256 + t;          // 0..4095 = (c, n)
            int c = idx >> 5;
            int n = idx & 31;
            float py = sPy[k][n], px = sPx[k][n], m = sM[k][n];
            float y0f = floorf(py), x0f = floorf(px);
            int   y0  = (int)y0f,   x0 = (int)x0f;
            float wy = py - y0f, wx = px - x0f;
            const float* xp = x + (((size_t)b * CC + c) << 12);

            float v00 = 0.f, v01 = 0.f, v10 = 0.f, v11 = 0.f;
            bool yin0 = (y0 >= 0)     && (y0 < HH);
            bool yin1 = (y0 + 1 >= 0) && (y0 + 1 < HH);
            bool xin0 = (x0 >= 0)     && (x0 < WW);
            bool xin1 = (x0 + 1 >= 0) && (x0 + 1 < WW);
            if (yin0 && xin0) v00 = xp[(y0 << 6) + x0];
            if (yin0 && xin1) v01 = xp[(y0 << 6) + x0 + 1];
            if (yin1 && xin0) v10 = xp[((y0 + 1) << 6) + x0];
            if (yin1 && xin1) v11 = xp[((y0 + 1) << 6) + x0 + 1];
            float s = ((v00 * (1.f - wy) + v10 * wy) * (1.f - wx) +
                       (v01 * (1.f - wy) + v11 * wy) * wx) * m;

            // scatter into B-fragment slot
            int chunk = k * 4 + (c >> 5);
            int ntile = n >> 4;
            int nn    = n & 15;
            int K  = c & 31;
            int lane = nn + (((K >> 3) & 1) ? 16 : 0);
            int k2 = (K & 7) + ((K >> 4) << 3);
            int vv = k2 >> 1;
            int hb = k2 & 1;
            sB[(chunk * NT + ntile) * 512 + lane * 16 + vv * 2 + hb] = f2bf(s);
        }
    }
    __syncthreads();

    // ---- Phase 2: 36 chunks x 2 independent WMMA streams per wave ----
    const int wave = t >> 5;      // o-tile
    const int lane = t & 31;

    v8f acc0 = {};
    v8f acc1 = {};
    #pragma unroll 4
    for (int chunk = 0; chunk < NCH; ++chunk) {
        v16bf a  = *(const v16bf*)(wA + ((size_t)((chunk * OTIL + wave) * 32 + lane) << 4));
        v16bf b0 = *(const v16bf*)(&sB[(chunk * NT + 0) * 512 + lane * 16]);
        v16bf b1 = *(const v16bf*)(&sB[(chunk * NT + 1) * 512 + lane * 16]);
        acc0 = __builtin_amdgcn_wmma_f32_16x16x32_bf16(
                   false, a, false, b0, (short)0, acc0, false, false);
        acc1 = __builtin_amdgcn_wmma_f32_16x16x32_bf16(
                   false, a, false, b1, (short)0, acc1, false, false);
    }

    // ---- Store D (C/D layout: VGPR i -> M=i (lanes 0-15) / M=i+8) ----
    int nloc = lane & 15;
    int mofs = (lane >> 4) * 8;
    int o_base = wave * 16;
    #pragma unroll
    for (int i = 0; i < 8; ++i) {
        int o = o_base + mofs + i;
        float bi = bias[o];
        size_t rowbase = (((size_t)b * OO + o) * HH + h) * WW;
        out[rowbase + w0 + nloc]      = acc0[i] + bi;
        out[rowbase + w0 + 16 + nloc] = acc1[i] + bi;
    }
}

// ------------------------------------------------------------------
extern "C" void kernel_launch(void* const* d_in, const int* in_sizes, int n_in,
                              void* d_out, int out_size, void* d_ws, size_t ws_size,
                              hipStream_t stream) {
    const float* x      = (const float*)d_in[0];
    const float* w_off  = (const float*)d_in[1];
    const float* b_off  = (const float*)d_in[2];
    const float* w_mask = (const float*)d_in[3];
    const float* b_mask = (const float*)d_in[4];
    const float* weight = (const float*)d_in[5];
    const float* bias   = (const float*)d_in[6];
    float* out = (float*)d_out;

    // workspace: offm (4*27*64*64 f32 = 1,769,472 B) then packed bf16 weights (294,912 B)
    float* offm = (float*)d_ws;
    unsigned int* wA = (unsigned int*)((char*)d_ws + (size_t)BB * 27 * HH * WW * sizeof(float));

    {   // offsets + masks
        int total = BB * 27 * HH * WW;
        offmask_kernel<<<(total + 255) / 256, 256, 0, stream>>>(
            x, w_off, b_off, w_mask, b_mask, offm);
    }
    {   // weight repack to A-fragment bf16
        int total = NCH * OTIL * 32 * 8;
        packw_kernel<<<(total + 255) / 256, 256, 0, stream>>>(weight, wA);
    }
    {   // main deformable conv via WMMA
        dim3 grid(WW / 32, HH, BB);
        deform_wmma_kernel<<<grid, 256, 0, stream>>>(
            x, offm, (const unsigned short*)wA, bias, out);
    }
}
